// QuaternionTransformer_7267084665369
// MI455X (gfx1250) — matrix operand order repaired and checked by reading, not compiled, not run on hardware
//
#include <hip/hip_runtime.h>
#include <cmath>

// ---------------------------------------------------------------------------
// Quaternion Transformer on MI455X (gfx1250).
// Strategy: expand every quaternion linear into a real bf16 GEMM
// [4*Din x 4*Dout] and drive everything through one WMMA bf16 GEMM kernel
// (v_wmma_f32_16x16x32_bf16, wave32, 128x128x32 block tiles, LDS staged via
// CDNA5 async global->LDS loads, drained with s_wait_asynccnt).
// ---------------------------------------------------------------------------

typedef __attribute__((ext_vector_type(16))) __bf16 v16bf;
typedef __attribute__((ext_vector_type(8)))  __bf16 v8bf;
typedef __attribute__((ext_vector_type(8)))  float  v8f;

namespace qtcfg {
constexpr int Vv = 32000, Dd = 256, Hf = 1024, NHh = 8, LEn = 2, LDn = 2;
constexpr int Bn = 4, Tn = 512;
constexpr int BT = Bn * Tn;        // 2048 token rows
constexpr int DQ = Dd * 4;         // 1024 real dims of the residual stream
constexpr int HQ = Hf * 4;         // 4096 real dims of the FF hidden
constexpr int HW = (Dd / NHh) * 4; // 128 real dims per head
}

// Hamilton-product expansion tables, indexed [cout][cin].
__constant__ int   qt_widx[4][4] = {{0,1,2,3},{1,0,3,2},{2,3,0,1},{3,2,1,0}};
__constant__ float qt_sign[4][4] = {{1.f,-1.f,-1.f,-1.f},
                                    {1.f, 1.f, 1.f,-1.f},
                                    {1.f,-1.f, 1.f, 1.f},
                                    {1.f, 1.f,-1.f, 1.f}};

// ---------------------------------------------------------------------------
// Weight expansion: W[4, Dout, Din] (f32) -> M[4*Din, 4*Dout] (bf16), so that
// row (din*4+cin), col (dout*4+cout) carries sign*W[widx][dout][din].
// ---------------------------------------------------------------------------
__global__ void qt_expand_kernel(const float* __restrict__ W,
                                 __bf16* __restrict__ Mx,
                                 int Din, int Dout) {
    const int cols = 4 * Dout;
    const int idx  = blockIdx.x * blockDim.x + threadIdx.x;
    if (idx >= 4 * Din * cols) return;
    const int row = idx / cols, col = idx % cols;
    const int din = row >> 2, cin = row & 3;
    const int dout = col >> 2, cout = col & 3;
    const float v = qt_sign[cout][cin] *
        W[(long long)qt_widx[cout][cin] * Dout * Din + (long long)dout * Din + din];
    Mx[idx] = (__bf16)v;
}

__global__ void qt_cast_bf16_kernel(const float* __restrict__ src,
                                    __bf16* __restrict__ dst, int n) {
    const int i = blockIdx.x * blockDim.x + threadIdx.x;
    if (i < n) dst[i] = (__bf16)src[i];
}

// ---------------------------------------------------------------------------
// Embedding: gather + quaternion normalize + RoPE-style angle scaling.
// Writes f32 residual stream and bf16 GEMM copy.
// ---------------------------------------------------------------------------
__global__ void qt_embed_kernel(const int* __restrict__ ids,
                                const float* __restrict__ emb,
                                float* __restrict__ X32,
                                __bf16* __restrict__ X16) {
    using namespace qtcfg;
    const int idx = blockIdx.x * blockDim.x + threadIdx.x;
    if (idx >= BT * Dd) return;
    const int bt = idx / Dd, d = idx % Dd;
    const int t  = bt % Tn;
    const int id = ids[bt];
    const long long VD = (long long)Vv * Dd;
    const float sc = __powf(10000.f, -(float)d / (float)Dd);
    const float r = emb[(long long)id * Dd + d];
    const float i = emb[VD     + (long long)id * Dd + d] * sc;
    const float j = emb[2 * VD + (long long)id * Dd + d] * sc;
    const float k = emb[3 * VD + (long long)id * Dd + d] * sc;
    const float n = sqrtf(r * r + i * i + j * j + k * k + 1e-6f);
    float ang;
    if (d < Dd / 2) ang = __sinf((float)t * __powf(10000.f, -(2.f * d) / (float)Dd));
    else            ang = __cosf((float)t * __powf(10000.f, -(2.f * (d - Dd / 2)) / (float)Dd));
    const float f = ang / n;
    const long long base = (long long)bt * DQ + d * 4;
    const float o0 = r * f, o1 = i * f, o2 = j * f, o3 = k * f;
    X32[base + 0] = o0; X32[base + 1] = o1; X32[base + 2] = o2; X32[base + 3] = o3;
    X16[base + 0] = (__bf16)o0; X16[base + 1] = (__bf16)o1;
    X16[base + 2] = (__bf16)o2; X16[base + 3] = (__bf16)o3;
}

// ---------------------------------------------------------------------------
// residual add + quaternion norm (qnorm): x = g*(v/||v||) + beta
// ---------------------------------------------------------------------------
__global__ void qt_resqnorm_kernel(float* __restrict__ X32,
                                   const float* __restrict__ Y,
                                   const float* __restrict__ g,
                                   const float* __restrict__ be,
                                   __bf16* __restrict__ X16) {
    using namespace qtcfg;
    const int idx = blockIdx.x * blockDim.x + threadIdx.x;
    if (idx >= BT * Dd) return;
    const int d = idx % Dd;
    const long long base = (long long)idx * 4;
    float v[4], s = 0.f;
    #pragma unroll
    for (int c = 0; c < 4; ++c) { v[c] = X32[base + c] + Y[base + c]; s += v[c] * v[c]; }
    const float inv = 1.f / sqrtf(s + 1e-6f);
    #pragma unroll
    for (int c = 0; c < 4; ++c) {
        const float o = g[d * 4 + c] * v[c] * inv + be[d * 4 + c];
        X32[base + c] = o;
        X16[base + c] = (__bf16)o;
    }
}

// modrelu: scale quaternion by relu(||v||+b)/(||v||+1e-6), emit bf16.
__global__ void qt_modrelu_kernel(const float* __restrict__ Y,
                                  const float* __restrict__ mb,
                                  __bf16* __restrict__ O) {
    using namespace qtcfg;
    const int idx = blockIdx.x * blockDim.x + threadIdx.x;
    if (idx >= BT * Hf) return;
    const int h = idx % Hf;
    const long long base = (long long)idx * 4;
    float s = 0.f;
    #pragma unroll
    for (int c = 0; c < 4; ++c) { const float v = Y[base + c]; s += v * v; }
    const float nrm = sqrtf(s);
    const float sc  = fmaxf(nrm + mb[h], 0.f) / (nrm + 1e-6f);
    #pragma unroll
    for (int c = 0; c < 4; ++c) O[base + c] = (__bf16)(Y[base + c] * sc);
}

// quaternion magnitude -> real features for the vocab head
__global__ void qt_realnorm_kernel(const float* __restrict__ X32,
                                   __bf16* __restrict__ R) {
    using namespace qtcfg;
    const int idx = blockIdx.x * blockDim.x + threadIdx.x;
    if (idx >= BT * Dd) return;
    const long long base = (long long)idx * 4;
    float s = 0.f;
    #pragma unroll
    for (int c = 0; c < 4; ++c) { const float v = X32[base + c]; s += v * v; }
    R[idx] = (__bf16)sqrtf(s);
}

// row softmax (optionally causal), bf16 output probabilities
__global__ __launch_bounds__(256) void qt_softmax_kernel(
        const float* __restrict__ S, __bf16* __restrict__ P,
        int Tlen, int causal) {
    const long long row = blockIdx.x;
    const int q   = (int)(row % Tlen);
    const int lim = causal ? q : (Tlen - 1);
    const float* s = S + row * Tlen;
    __bf16* p = P + row * Tlen;
    __shared__ float red[256];
    const int tid = threadIdx.x;
    float mx = -3.4e38f;
    for (int j = tid; j < Tlen; j += 256)
        if (j <= lim) mx = fmaxf(mx, s[j]);
    red[tid] = mx; __syncthreads();
    for (int st = 128; st > 0; st >>= 1) {
        if (tid < st) red[tid] = fmaxf(red[tid], red[tid + st]);
        __syncthreads();
    }
    mx = red[0]; __syncthreads();
    float sum = 0.f;
    for (int j = tid; j < Tlen; j += 256)
        if (j <= lim) sum += __expf(s[j] - mx);
    red[tid] = sum; __syncthreads();
    for (int st = 128; st > 0; st >>= 1) {
        if (tid < st) red[tid] += red[tid + st];
        __syncthreads();
    }
    const float inv = 1.f / red[0];
    for (int j = tid; j < Tlen; j += 256)
        p[j] = (__bf16)((j <= lim) ? __expf(s[j] - mx) * inv : 0.f);
}

// ---------------------------------------------------------------------------
// CDNA5 async global->LDS staging helper: copies 32 contiguous bytes from a
// 16B-aligned global address directly into LDS (no VGPR round trip).
// Tracked by ASYNCcnt; drain with qt_wait_async() before the consume barrier.
// ---------------------------------------------------------------------------
__device__ __forceinline__ void qt_async_copy32(const __bf16* gptr, __bf16* lptr) {
    const unsigned           lds = (unsigned)(size_t)lptr;            // LDS byte offset
    const unsigned long long ga  = (unsigned long long)(size_t)gptr;  // global address
    asm volatile(
        "global_load_async_to_lds_b128 %0, %1, off\n\t"
        "global_load_async_to_lds_b128 %0, %1, off offset:16"
        :: "v"(lds), "v"(ga) : "memory");
}
__device__ __forceinline__ void qt_wait_async() {
    asm volatile("s_wait_asynccnt 0x0" ::: "memory");
}

// ---------------------------------------------------------------------------
// WMMA bf16 GEMM.  C[M,N](f32 or bf16) = alpha * A[M,K] * B(+bias).
// Block tile 128x128, BK=32, 256 threads = 8 wave32 waves in a 2x4 grid,
// each wave computes a 64x32 patch = 4x2 v_wmma_f32_16x16x32_bf16 tiles.
// A (and B when already [N][K]) are staged with async global->LDS b128 ops;
// the [K][N] B path is scatter-transposed through VGPRs. B is held
// transposed in LDS so all fragment loads are ds_load_b128.
// Batched via blockIdx.z with split strides: off = (z/zdiv)*s0 + (z%zdiv)*s1.
// All M,N multiples of 128 and K multiples of 32 by construction.
// ---------------------------------------------------------------------------
template <int TRANSB, int OUTBF16>
__global__ __launch_bounds__(256) void qt_gemm_wmma(
        const __bf16* __restrict__ A, const __bf16* __restrict__ Bm,
        void* __restrict__ Cv, const float* __restrict__ bias,
        int M, int N, int K, int lda, int ldb, int ldc,
        long long sA0, long long sA1, long long sB0, long long sB1,
        long long sC0, long long sC1, int zdiv, float alpha) {
    constexpr int LS = 40;  // LDS row stride in halfs: 80B -> conflict-friendly, 16B aligned
    __shared__ __align__(16) __bf16 As[128 * LS];
    __shared__ __align__(16) __bf16 Bs[128 * LS];

    const int z = blockIdx.z;
    A  += (long long)(z / zdiv) * sA0 + (long long)(z % zdiv) * sA1;
    Bm += (long long)(z / zdiv) * sB0 + (long long)(z % zdiv) * sB1;
    const long long coff = (long long)(z / zdiv) * sC0 + (long long)(z % zdiv) * sC1;

    const int m0 = blockIdx.y * 128, n0 = blockIdx.x * 128;
    const int tid = threadIdx.x, lane = tid & 31, wave = tid >> 5;
    const int wm = wave >> 2;  // 0..1 -> 64-row strip
    const int wn = wave & 3;   // 0..3 -> 32-col strip

    const v8f vzero = {0.f, 0.f, 0.f, 0.f, 0.f, 0.f, 0.f, 0.f};
    v8f acc[4][2];
    #pragma unroll
    for (int i = 0; i < 4; ++i)
        #pragma unroll
        for (int j = 0; j < 2; ++j) acc[i][j] = vzero;

    union Frag { v16bf v; v8bf h[2]; };

    const int nk = K >> 5;
    for (int kt = 0; kt < nk; ++kt) {
        const int k0 = kt << 5;
        __syncthreads();
        // ---- stage A: 128x32 row-major (async global->LDS, 32B/lane) ----
        {
            const int r = tid >> 1, off = (tid & 1) << 4;
            qt_async_copy32(A + (long long)(m0 + r) * lda + k0 + off,
                            &As[r * LS + off]);
        }
        // ---- stage B transposed: Bs[n][k] ----
        if (TRANSB) {  // source already [N][K]: async direct to LDS
            const int n = tid >> 1, off = (tid & 1) << 4;
            qt_async_copy32(Bm + (long long)(n0 + n) * ldb + k0 + off,
                            &Bs[n * LS + off]);
        } else {       // source [K][N]; scatter-transpose through VGPRs
            const int k = tid >> 3, nn = (tid & 7) << 4;
            const v8bf* g = (const v8bf*)(Bm + (long long)(k0 + k) * ldb + n0 + nn);
            const v8bf x0 = g[0], x1 = g[1];
            #pragma unroll
            for (int j = 0; j < 8; ++j) {
                Bs[(nn + j) * LS + k]     = x0[j];
                Bs[(nn + 8 + j) * LS + k] = x1[j];
            }
        }
        if (kt + 1 < nk)  // global_prefetch_b8 the next K tile of A into L2
            __builtin_prefetch(A + (long long)(m0 + (tid >> 1)) * lda + k0 + 32, 0, 1);
        qt_wait_async();  // drain ASYNCcnt before making the tile visible
        __syncthreads();

        // ---- fragments + WMMA (ISA 7.12.2 wave32 layouts) ----
        const int g8  = (lane >> 4) << 3;  // A: K offset 0/8 by lane group
        const int g16 = (lane >> 4) << 4;  // unused; B fragments use g8 (0/8) like A
        const int ml  = lane & 15;

        Frag bf[2];
        #pragma unroll
        for (int nt = 0; nt < 2; ++nt) {
            const int n = wn * 32 + nt * 16 + ml;
            bf[nt].h[0] = *(const v8bf*)&Bs[n * LS + g8];
            bf[nt].h[1] = *(const v8bf*)&Bs[n * LS + 16 + g8];
        }
        #pragma unroll
        for (int mt = 0; mt < 4; ++mt) {
            const int m = wm * 64 + mt * 16 + ml;
            Frag af;
            af.h[0] = *(const v8bf*)&As[m * LS + g8];
            af.h[1] = *(const v8bf*)&As[m * LS + 16 + g8];
            #pragma unroll
            for (int nt = 0; nt < 2; ++nt)
                acc[mt][nt] = __builtin_amdgcn_wmma_f32_16x16x32_bf16(
                    false, af.v, false, bf[nt].v, (short)0, acc[mt][nt], false, false);
        }
    }

    // ---- epilogue: alpha, bias, store f32 or bf16 ----
    const int rowg = (lane >> 4) << 3;
    const int coll = lane & 15;
    #pragma unroll
    for (int mt = 0; mt < 4; ++mt)
        #pragma unroll
        for (int nt = 0; nt < 2; ++nt) {
            const int col = n0 + wn * 32 + nt * 16 + coll;
            const float badd = bias ? bias[col] : 0.f;
            #pragma unroll
            for (int r = 0; r < 8; ++r) {
                const int row = m0 + wm * 64 + mt * 16 + rowg + r;
                const float v = acc[mt][nt][r] * alpha + badd;
                if (OUTBF16)
                    ((__bf16*)Cv)[coff + (long long)row * ldc + col] = (__bf16)v;
                else
                    ((float*)Cv)[coff + (long long)row * ldc + col] = v;
            }
        }
}

// ---------------------------------------------------------------------------
// Host-side orchestration
// ---------------------------------------------------------------------------
static void launch_gemm(hipStream_t st, const __bf16* A, const __bf16* Bm, void* C,
                        const float* bias, int M, int N, int K,
                        int lda, int ldb, int ldc,
                        long long sA0, long long sA1, long long sB0, long long sB1,
                        long long sC0, long long sC1, int nz, int zdiv,
                        float alpha, bool transB, bool outBF16) {
    dim3 grid((unsigned)(N / 128), (unsigned)(M / 128), (unsigned)nz), block(256);
    if (transB) {
        if (outBF16) qt_gemm_wmma<1, 1><<<grid, block, 0, st>>>(A, Bm, C, bias, M, N, K, lda, ldb, ldc, sA0, sA1, sB0, sB1, sC0, sC1, zdiv, alpha);
        else         qt_gemm_wmma<1, 0><<<grid, block, 0, st>>>(A, Bm, C, bias, M, N, K, lda, ldb, ldc, sA0, sA1, sB0, sB1, sC0, sC1, zdiv, alpha);
    } else {
        if (outBF16) qt_gemm_wmma<0, 1><<<grid, block, 0, st>>>(A, Bm, C, bias, M, N, K, lda, ldb, ldc, sA0, sA1, sB0, sB1, sC0, sC1, zdiv, alpha);
        else         qt_gemm_wmma<0, 0><<<grid, block, 0, st>>>(A, Bm, C, bias, M, N, K, lda, ldb, ldc, sA0, sA1, sB0, sB1, sC0, sC1, zdiv, alpha);
    }
}

// Full attention block (projections, scores, softmax, context, out-proj).
static void run_attn(hipStream_t st, const __bf16* qin, const __bf16* kvin,
                     const __bf16* Wq, const __bf16* Wk, const __bf16* Wv, const __bf16* Wo,
                     const float* bias4, bool causal,
                     __bf16* Qb, __bf16* Kb, __bf16* Vb,
                     float* Sc, __bf16* Pr, __bf16* Ob, float* Yout) {
    using namespace qtcfg;
    launch_gemm(st, qin,  Wq, Qb, bias4 + 0 * DQ, BT, DQ, DQ, DQ, DQ, DQ, 0,0,0,0,0,0, 1, 1, 1.f, false, true);
    launch_gemm(st, kvin, Wk, Kb, bias4 + 1 * DQ, BT, DQ, DQ, DQ, DQ, DQ, 0,0,0,0,0,0, 1, 1, 1.f, false, true);
    launch_gemm(st, kvin, Wv, Vb, bias4 + 2 * DQ, BT, DQ, DQ, DQ, DQ, DQ, 0,0,0,0,0,0, 1, 1, 1.f, false, true);
    // scores[b,h] = Q_head * K_head^T / sqrt(128)
    launch_gemm(st, Qb, Kb, Sc, nullptr, Tn, Tn, HW, DQ, DQ, Tn,
                (long long)Tn * DQ, HW, (long long)Tn * DQ, HW,
                (long long)NHh * Tn * Tn, (long long)Tn * Tn,
                Bn * NHh, NHh, 1.f / sqrtf((float)HW), true, false);
    qt_softmax_kernel<<<Bn * NHh * Tn, 256, 0, st>>>(Sc, Pr, Tn, causal ? 1 : 0);
    // context[b,h] = P * V_head  (merged back by column offset h*128)
    launch_gemm(st, Pr, Vb, Ob, nullptr, Tn, HW, Tn, Tn, DQ, DQ,
                (long long)NHh * Tn * Tn, (long long)Tn * Tn,
                (long long)Tn * DQ, HW, (long long)Tn * DQ, HW,
                Bn * NHh, NHh, 1.f, false, true);
    launch_gemm(st, Ob, Wo, Yout, bias4 + 3 * DQ, BT, DQ, DQ, DQ, DQ, DQ, 0,0,0,0,0,0, 1, 1, 1.f, false, false);
}

extern "C" void kernel_launch(void* const* d_in, const int* in_sizes, int n_in,
                              void* d_out, int out_size, void* d_ws, size_t ws_size,
                              hipStream_t stream) {
    using namespace qtcfg;
    (void)in_sizes; (void)n_in; (void)out_size; (void)ws_size;

    const int*   src        = (const int*)d_in[0];
    const int*   tgt        = (const int*)d_in[1];
    const float* emb        = (const float*)d_in[3];
    const float* enc_attn_W = (const float*)d_in[4];
    const float* enc_attn_b = (const float*)d_in[5];
    const float* enc_norm_g = (const float*)d_in[6];
    const float* enc_norm_b = (const float*)d_in[7];
    const float* enc_ff_W1  = (const float*)d_in[8];
    const float* enc_ff_b1  = (const float*)d_in[9];
    const float* enc_ff_mod = (const float*)d_in[10];
    const float* enc_ff_W2  = (const float*)d_in[11];
    const float* enc_ff_b2  = (const float*)d_in[12];
    const float* dec_sa_W   = (const float*)d_in[13];
    const float* dec_sa_b   = (const float*)d_in[14];
    const float* dec_ca_W   = (const float*)d_in[15];
    const float* dec_ca_b   = (const float*)d_in[16];
    const float* dec_norm_g = (const float*)d_in[17];
    const float* dec_norm_b = (const float*)d_in[18];
    const float* dec_ff_W1  = (const float*)d_in[19];
    const float* dec_ff_b1  = (const float*)d_in[20];
    const float* dec_ff_mod = (const float*)d_in[21];
    const float* dec_ff_W2  = (const float*)d_in[22];
    const float* dec_ff_b2  = (const float*)d_in[23];
    const float* fc_W       = (const float*)d_in[24];
    const float* fc_b       = (const float*)d_in[25];

    // ---- workspace bump allocator ----
    size_t off = 0;
    char* base = (char*)d_ws;
    auto alloc = [&](size_t bytes) -> char* {
        off = (off + 255) & ~(size_t)255;
        char* p = base + off;
        off += bytes;
        return p;
    };

    // expanded bf16 weights
    __bf16* encWe[LEn][4]; __bf16* saWe[LDn][4]; __bf16* caWe[LDn][4];
    __bf16* encW1e[LEn]; __bf16* encW2e[LEn]; __bf16* decW1e[LDn]; __bf16* decW2e[LDn];
    for (int l = 0; l < LEn; ++l) for (int p = 0; p < 4; ++p) encWe[l][p] = (__bf16*)alloc((size_t)DQ * DQ * 2);
    for (int l = 0; l < LDn; ++l) for (int p = 0; p < 4; ++p) saWe[l][p]  = (__bf16*)alloc((size_t)DQ * DQ * 2);
    for (int l = 0; l < LDn; ++l) for (int p = 0; p < 4; ++p) caWe[l][p]  = (__bf16*)alloc((size_t)DQ * DQ * 2);
    for (int l = 0; l < LEn; ++l) { encW1e[l] = (__bf16*)alloc((size_t)DQ * HQ * 2); encW2e[l] = (__bf16*)alloc((size_t)HQ * DQ * 2); }
    for (int l = 0; l < LDn; ++l) { decW1e[l] = (__bf16*)alloc((size_t)DQ * HQ * 2); decW2e[l] = (__bf16*)alloc((size_t)HQ * DQ * 2); }
    __bf16* fcWb = (__bf16*)alloc((size_t)Vv * Dd * 2);

    // activations
    float*  Xs32 = (float*)alloc((size_t)BT * DQ * 4);   // encoder residual (f32)
    __bf16* Xs16 = (__bf16*)alloc((size_t)BT * DQ * 2);
    float*  Xt32 = (float*)alloc((size_t)BT * DQ * 4);   // decoder residual (f32)
    __bf16* Xt16 = (__bf16*)alloc((size_t)BT * DQ * 2);
    __bf16* Qb   = (__bf16*)alloc((size_t)BT * DQ * 2);
    __bf16* Kb   = (__bf16*)alloc((size_t)BT * DQ * 2);
    __bf16* Vb   = (__bf16*)alloc((size_t)BT * DQ * 2);
    float*  Sc   = (float*)alloc((size_t)Bn * NHh * Tn * Tn * 4);
    __bf16* Pr   = (__bf16*)alloc((size_t)Bn * NHh * Tn * Tn * 2);
    __bf16* Ob   = (__bf16*)alloc((size_t)BT * DQ * 2);
    float*  Yf   = (float*)alloc((size_t)BT * HQ * 4);   // shared f32 GEMM output
    __bf16* T1b  = (__bf16*)alloc((size_t)BT * HQ * 2);
    __bf16* Rb   = (__bf16*)alloc((size_t)BT * Dd * 2);

    const dim3 blk(256);
    auto ceil256 = [](long long n) { return (unsigned)((n + 255) / 256); };

    // ---- 1) expand all quaternion weights to real bf16 GEMM matrices ----
    for (int l = 0; l < LEn; ++l)
        for (int p = 0; p < 4; ++p)
            qt_expand_kernel<<<ceil256((long long)DQ * DQ), blk, 0, stream>>>(
                enc_attn_W + (long long)(l * 4 + p) * 4 * Dd * Dd, encWe[l][p], Dd, Dd);
    for (int l = 0; l < LDn; ++l)
        for (int p = 0; p < 4; ++p) {
            qt_expand_kernel<<<ceil256((long long)DQ * DQ), blk, 0, stream>>>(
                dec_sa_W + (long long)(l * 4 + p) * 4 * Dd * Dd, saWe[l][p], Dd, Dd);
            qt_expand_kernel<<<ceil256((long long)DQ * DQ), blk, 0, stream>>>(
                dec_ca_W + (long long)(l * 4 + p) * 4 * Dd * Dd, caWe[l][p], Dd, Dd);
        }
    for (int l = 0; l < LEn; ++l) {
        qt_expand_kernel<<<ceil256((long long)DQ * HQ), blk, 0, stream>>>(
            enc_ff_W1 + (long long)l * 4 * Hf * Dd, encW1e[l], Dd, Hf);
        qt_expand_kernel<<<ceil256((long long)HQ * DQ), blk, 0, stream>>>(
            enc_ff_W2 + (long long)l * 4 * Dd * Hf, encW2e[l], Hf, Dd);
    }
    for (int l = 0; l < LDn; ++l) {
        qt_expand_kernel<<<ceil256((long long)DQ * HQ), blk, 0, stream>>>(
            dec_ff_W1 + (long long)l * 4 * Hf * Dd, decW1e[l], Dd, Hf);
        qt_expand_kernel<<<ceil256((long long)HQ * DQ), blk, 0, stream>>>(
            dec_ff_W2 + (long long)l * 4 * Dd * Hf, decW2e[l], Hf, Dd);
    }
    qt_cast_bf16_kernel<<<ceil256((long long)Vv * Dd), blk, 0, stream>>>(fc_W, fcWb, Vv * Dd);

    // ---- 2) embeddings + rope ----
    qt_embed_kernel<<<ceil256((long long)BT * Dd), blk, 0, stream>>>(src, emb, Xs32, Xs16);
    qt_embed_kernel<<<ceil256((long long)BT * Dd), blk, 0, stream>>>(tgt, emb, Xt32, Xt16);

    // ---- 3) encoder stack ----
    for (int l = 0; l < LEn; ++l) {
        run_attn(stream, Xs16, Xs16, encWe[l][0], encWe[l][1], encWe[l][2], encWe[l][3],
                 enc_attn_b + (long long)l * 4 * DQ, false, Qb, Kb, Vb, Sc, Pr, Ob, Yf);
        qt_resqnorm_kernel<<<ceil256((long long)BT * Dd), blk, 0, stream>>>(
            Xs32, Yf, enc_norm_g + (long long)(l * 2 + 0) * DQ,
            enc_norm_b + (long long)(l * 2 + 0) * DQ, Xs16);
        launch_gemm(stream, Xs16, encW1e[l], Yf, enc_ff_b1 + (long long)l * HQ,
                    BT, HQ, DQ, DQ, HQ, HQ, 0,0,0,0,0,0, 1, 1, 1.f, false, false);
        qt_modrelu_kernel<<<ceil256((long long)BT * Hf), blk, 0, stream>>>(
            Yf, enc_ff_mod + (long long)l * Hf, T1b);
        launch_gemm(stream, T1b, encW2e[l], Yf, enc_ff_b2 + (long long)l * DQ,
                    BT, DQ, HQ, HQ, DQ, DQ, 0,0,0,0,0,0, 1, 1, 1.f, false, false);
        qt_resqnorm_kernel<<<ceil256((long long)BT * Dd), blk, 0, stream>>>(
            Xs32, Yf, enc_norm_g + (long long)(l * 2 + 1) * DQ,
            enc_norm_b + (long long)(l * 2 + 1) * DQ, Xs16);
    }
    // Xs16/Xs32 now hold encoder memory `m`.

    // ---- 4) decoder stack ----
    for (int l = 0; l < LDn; ++l) {
        // causal self-attention
        run_attn(stream, Xt16, Xt16, saWe[l][0], saWe[l][1], saWe[l][2], saWe[l][3],
                 dec_sa_b + (long long)l * 4 * DQ, true, Qb, Kb, Vb, Sc, Pr, Ob, Yf);
        qt_resqnorm_kernel<<<ceil256((long long)BT * Dd), blk, 0, stream>>>(
            Xt32, Yf, dec_norm_g + (long long)(l * 3 + 0) * DQ,
            dec_norm_b + (long long)(l * 3 + 0) * DQ, Xt16);
        // cross-attention against encoder memory
        run_attn(stream, Xt16, Xs16, caWe[l][0], caWe[l][1], caWe[l][2], caWe[l][3],
                 dec_ca_b + (long long)l * 4 * DQ, false, Qb, Kb, Vb, Sc, Pr, Ob, Yf);
        qt_resqnorm_kernel<<<ceil256((long long)BT * Dd), blk, 0, stream>>>(
            Xt32, Yf, dec_norm_g + (long long)(l * 3 + 1) * DQ,
            dec_norm_b + (long long)(l * 3 + 1) * DQ, Xt16);
        // feed-forward
        launch_gemm(stream, Xt16, decW1e[l], Yf, dec_ff_b1 + (long long)l * HQ,
                    BT, HQ, DQ, DQ, HQ, HQ, 0,0,0,0,0,0, 1, 1, 1.f, false, false);
        qt_modrelu_kernel<<<ceil256((long long)BT * Hf), blk, 0, stream>>>(
            Yf, dec_ff_mod + (long long)l * Hf, T1b);
        launch_gemm(stream, T1b, decW2e[l], Yf, dec_ff_b2 + (long long)l * DQ,
                    BT, DQ, HQ, HQ, DQ, DQ, 0,0,0,0,0,0, 1, 1, 1.f, false, false);
        qt_resqnorm_kernel<<<ceil256((long long)BT * Dd), blk, 0, stream>>>(
            Xt32, Yf, dec_norm_g + (long long)(l * 3 + 2) * DQ,
            dec_norm_b + (long long)(l * 3 + 2) * DQ, Xt16);
    }

    // ---- 5) quaternion magnitude + vocab head ----
    qt_realnorm_kernel<<<ceil256((long long)BT * Dd), blk, 0, stream>>>(Xt32, Rb);
    launch_gemm(stream, Rb, fcWb, (float*)d_out, fc_b,
                BT, Vv, Dd, Dd, Dd, Vv, 0,0,0,0,0,0, 1, 1, 1.f, true, false);
}
